// TorchLMHeadCPO_48859547959893
// MI455X (gfx1250) — matrix operand order repaired and checked by reading, not compiled, not run on hardware
//
#include <hip/hip_runtime.h>
#include <hip/hip_bf16.h>
#include <math.h>

// ---------------- problem constants (match reference) ----------------
constexpr int kB2 = 8;            // 2*B sequences
constexpr int kT  = 512;          // tokens per sequence
constexpr int kH  = 4096;         // hidden
constexpr int kV  = 32000;        // vocab
constexpr int kNTok = kB2 * kT;   // 4096 tokens total

constexpr int kMTile   = 32;               // tokens per block
constexpr int kTokTiles = kNTok / kMTile;  // 128
constexpr int kVChunk  = 1280;             // vocab per block
constexpr int kNV      = kV / kVChunk;     // 25 chunks
constexpr int kOuters  = kVChunk / 128;    // 10 (8 waves * 16 cols)

constexpr float kBeta = 0.1f;
constexpr float kAlpha = 1.0f;
constexpr float kLabelSmooth = 0.0f;
constexpr int   kIgnore = -100;
constexpr float kNegInf = -3.4e38f;

typedef __attribute__((ext_vector_type(16))) __bf16          v16bf;
typedef __attribute__((ext_vector_type(8)))  unsigned int    v8u;
typedef __attribute__((ext_vector_type(8)))  float           v8f;

// Pack two f32 -> packed bf16 dword with one VALU op (v_cvt_pk_bf16_f32).
__device__ __forceinline__ unsigned int f2bf2(float lo, float hi) {
  unsigned int r;
  asm("v_cvt_pk_bf16_f32 %0, %1, %2" : "=v"(r) : "v"(lo), "v"(hi));
  return r;
}

__device__ __forceinline__ float hmax16(float v) {
#pragma unroll
  for (int i = 1; i < 16; i <<= 1) v = fmaxf(v, __shfl_xor(v, i, 16));
  return v;
}
__device__ __forceinline__ float hsum16(float v) {
#pragma unroll
  for (int i = 1; i < 16; i <<= 1) v += __shfl_xor(v, i, 16);
  return v;
}

// =====================================================================
// Pass 1: fused tiled GEMM (bf16 WMMA, fp32 accum) + online logsumexp
// grid = (kTokTiles, kNV); token tile is the fast axis so concurrent
// blocks share the same 20MB W vocab-chunk in L2.
// Software pipeline: global loads (regs) -> WMMA from LDS[p] ->
// convert+store regs into LDS[p^1] -> barrier.
// =====================================================================
__global__ __launch_bounds__(256) void cpo_partial(
    const float* __restrict__ x, const int* __restrict__ y,
    const float* __restrict__ W,
    float* __restrict__ pM, float* __restrict__ pS, float* __restrict__ pL)
{
  // double-buffered packed-bf16 tiles; row stride 17 dwords (pad) to
  // spread LDS banks. A: 32 rows x 16 dwords; B: 128 cols x 16 dwords.
  __shared__ unsigned int Asu[2][kMTile * 17];
  __shared__ unsigned int Bsu[2][128 * 17];
  __shared__ float Mrg[8][kMTile][3];

  const int tid  = threadIdx.x;
  const int wave = tid >> 5;
  const int lane = tid & 31;
  const int half = lane >> 4;     // which 16-lane half
  const int ln   = lane & 15;

  const int tokBase = blockIdx.x * kMTile;
  const int vchunk  = blockIdx.y;
  const int vBase   = vchunk * kVChunk;

  // this thread's fixed staging coordinates
  const int aM = (tid * 4) >> 5, aK = (tid * 4) & 31;   // A: 4 elems
  const int bN = tid >> 1, bK = (tid & 1) * 16;         // B: 16 elems

  // labels for the rows this lane covers (replicated across the 16-lane half)
  int   lab [2][8];
  float runM[2][8], runS[2][8], runL[2][8];
#pragma unroll
  for (int s = 0; s < 2; ++s)
#pragma unroll
    for (int r = 0; r < 8; ++r) {
      lab[s][r]  = y[tokBase + s * 16 + half * 8 + r];
      runM[s][r] = kNegInf;
      runS[s][r] = 0.f;
      runL[s][r] = 0.f;
    }

  // in-flight slab registers (held across the WMMA section)
  float4 la, lb0, lb1, lb2, lb3;

  auto loadSlab = [&](int k0, int colBase) {
    la = *reinterpret_cast<const float4*>(
        &x[(size_t)(tokBase + aM) * kH + k0 + aK]);
    const float4* wp = reinterpret_cast<const float4*>(
        &W[(size_t)(colBase + bN) * kH + k0 + bK]);
    lb0 = wp[0]; lb1 = wp[1]; lb2 = wp[2]; lb3 = wp[3];
    if (k0 + 32 < kH)   // uniform: prefetch the slab after this one
      __builtin_prefetch(&W[(size_t)(colBase + bN) * kH + k0 + 32 + bK], 0, 1);
  };
  auto storeSlab = [&](int p) {
    unsigned int* as = &Asu[p][aM * 17 + (aK >> 1)];
    as[0] = f2bf2(la.x, la.y);
    as[1] = f2bf2(la.z, la.w);
    unsigned int* bs = &Bsu[p][bN * 17 + (bK >> 1)];
    bs[0] = f2bf2(lb0.x, lb0.y); bs[1] = f2bf2(lb0.z, lb0.w);
    bs[2] = f2bf2(lb1.x, lb1.y); bs[3] = f2bf2(lb1.z, lb1.w);
    bs[4] = f2bf2(lb2.x, lb2.y); bs[5] = f2bf2(lb2.z, lb2.w);
    bs[6] = f2bf2(lb3.x, lb3.y); bs[7] = f2bf2(lb3.z, lb3.w);
  };

  for (int outer = 0; outer < kOuters; ++outer) {
    const int colBase = vBase + outer * 128;       // this iter: 128 vocab cols
    const int nGlob   = colBase + wave * 16 + ln;  // this lane's column
    v8f c0 = {};
    v8f c1 = {};

    // prologue: slab 0 -> buffer 0
    loadSlab(0, colBase);
    storeSlab(0);
    __syncthreads();

    for (int k0 = 0; k0 < kH; k0 += 32) {
      const int p = (k0 >> 5) & 1;
      const bool more = (k0 + 32 < kH);   // uniform
      if (more) loadSlab(k0 + 32, colBase);   // issue loads, consume later

      // Build fragments per CDNA5 16-bit WMMA layouts (dword-pair gathers).
      // A 16x32: lane -> M (lane&15); dword q -> K-pair index
      //          (q&3) + ((q>>2)<<3) + (half<<2)
      // B 32x16: lane -> N (lane&15); dword q -> K-pair same as A
      v8u a0u, a1u, bu;
#pragma unroll
      for (int q = 0; q < 8; ++q) {
        const int ka = (q & 3) + ((q >> 2) << 3) + (half << 2);
        a0u[q] = Asu[p][ln * 17 + ka];
        a1u[q] = Asu[p][(16 + ln) * 17 + ka];
        bu[q]  = Bsu[p][(wave * 16 + ln) * 17 + ka];
      }
      const v16bf a0 = __builtin_bit_cast(v16bf, a0u);
      const v16bf a1 = __builtin_bit_cast(v16bf, a1u);
      const v16bf b  = __builtin_bit_cast(v16bf, bu);
      c0 = __builtin_amdgcn_wmma_f32_16x16x32_bf16(false, a0, false, b,
                                                   (short)0, c0, false, false);
      c1 = __builtin_amdgcn_wmma_f32_16x16x32_bf16(false, a1, false, b,
                                                   (short)0, c1, false, false);

      if (more) storeSlab(p ^ 1);  // loadcnt wait lands here, after the WMMAs
      __syncthreads();             // staging of p^1 done; reads of p done
    }

    // Epilogue for this 32x16 logits fragment pair: online logsumexp per row.
    // C layout: element r, lanes 0-15 -> M=r, N=ln; lanes 16-31 -> M=8+r.
#pragma unroll
    for (int s = 0; s < 2; ++s) {
#pragma unroll
      for (int r = 0; r < 8; ++r) {
        const float val = (s == 0) ? c0[r] : c1[r];
        if (nGlob == lab[s][r]) runL[s][r] += val;   // label logit pick
        const float tmax = hmax16(val);
        const float nm = fmaxf(runM[s][r], tmax);
        const float es = hsum16(__expf(val - nm));
        runS[s][r] = runS[s][r] * __expf(runM[s][r] - nm) + es;
        runM[s][r] = nm;
      }
    }
  }

  // reduce label partials across the 16-lane half (only one lane matched)
#pragma unroll
  for (int s = 0; s < 2; ++s)
#pragma unroll
    for (int r = 0; r < 8; ++r) runL[s][r] = hsum16(runL[s][r]);

  if (ln == 0) {
#pragma unroll
    for (int s = 0; s < 2; ++s)
#pragma unroll
      for (int r = 0; r < 8; ++r) {
        const int row = s * 16 + half * 8 + r;
        Mrg[wave][row][0] = runM[s][r];
        Mrg[wave][row][1] = runS[s][r];
        Mrg[wave][row][2] = runL[s][r];
      }
  }
  __syncthreads();

  if (tid < kMTile) {  // merge the 8 waves' disjoint vocab columns
    float M = kNegInf, S = 0.f, L = 0.f;
#pragma unroll
    for (int w = 0; w < 8; ++w) {
      const float m2 = Mrg[w][tid][0];
      const float s2 = Mrg[w][tid][1];
      const float nm = fmaxf(M, m2);
      S = S * __expf(M - nm) + s2 * __expf(m2 - nm);
      M = nm;
      L += Mrg[w][tid][2];
    }
    const size_t o = (size_t)(tokBase + tid) * kNV + vchunk;
    pM[o] = M; pS[o] = S; pL[o] = L;
  }
}

// =====================================================================
// Pass 2: merge the kNV vocab-chunk partials per token
// =====================================================================
__global__ __launch_bounds__(256) void cpo_combine(
    const int* __restrict__ y,
    const float* __restrict__ pM, const float* __restrict__ pS,
    const float* __restrict__ pL, float* __restrict__ tokLogp)
{
  const int t = blockIdx.x * blockDim.x + threadIdx.x;
  if (t >= kNTok) return;
  float M = kNegInf, S = 0.f, L = 0.f;
  for (int c = 0; c < kNV; ++c) {
    const size_t o = (size_t)t * kNV + c;
    const float m2 = pM[o], s2 = pS[o];
    const float nm = fmaxf(M, m2);
    S = S * __expf(M - nm) + s2 * __expf(m2 - nm);
    M = nm;
    L += pL[o];
  }
  const float lse = M + __logf(S);
  tokLogp[t] = (y[t] != kIgnore) ? (L - lse) : 0.f;
}

// =====================================================================
// Pass 3: deterministic scalar reduction (fixed serial order, no atomics)
// =====================================================================
__device__ __forceinline__ float log_sigmoid(float z) {
  return (z < 0.f) ? (z - log1pf(__expf(z))) : (-log1pf(__expf(-z)));
}

__global__ void cpo_finalize(const int* __restrict__ y,
                             const float* __restrict__ tokLogp,
                             float* __restrict__ out)
{
  __shared__ float seq[kB2];
  __shared__ float nllNum, nllCnt;
  const int tid = threadIdx.x;
  if (tid < kB2) {
    float s = 0.f;
    for (int t = 0; t < kT; ++t) s += tokLogp[tid * kT + t];
    seq[tid] = s;
  }
  if (tid == kB2) {
    float num = 0.f, cnt = 0.f;
    for (int t = 0; t < (kB2 / 2) * kT; ++t)
      if (y[t] != kIgnore) { num += -tokLogp[t]; cnt += 1.f; }
    nllNum = num; nllCnt = cnt;
  }
  __syncthreads();
  if (tid == 0) {
    float acc = 0.f;
#pragma unroll
    for (int b = 0; b < kB2 / 2; ++b) {
      const float d = seq[b] - seq[b + kB2 / 2];
      const float z = kBeta * d;
      acc += -log_sigmoid(z) * (1.f - kLabelSmooth)
             - log_sigmoid(-z) * kLabelSmooth;
    }
    out[0] = (nllNum / fmaxf(nllCnt, 1.f)) * kAlpha + acc / (float)(kB2 / 2);
  }
}

// =====================================================================
extern "C" void kernel_launch(void* const* d_in, const int* in_sizes, int n_in,
                              void* d_out, int out_size, void* d_ws, size_t ws_size,
                              hipStream_t stream) {
  const float* x = (const float*)d_in[0];
  const int*   y = (const int*)d_in[1];   // integer input -> const int* per harness
  const float* W = (const float*)d_in[2];
  float* out = (float*)d_out;

  // workspace layout: 3 * (kNTok * kNV) partials + kNTok token logps (~1.25 MB)
  float* pM = (float*)d_ws;
  float* pS = pM + (size_t)kNTok * kNV;
  float* pL = pS + (size_t)kNTok * kNV;
  float* tokLogp = pL + (size_t)kNTok * kNV;
  (void)in_sizes; (void)n_in; (void)out_size; (void)ws_size;

  dim3 g1(kTokTiles, kNV);
  cpo_partial<<<g1, 256, 0, stream>>>(x, y, W, pM, pS, pL);
  cpo_combine<<<kNTok / 256, 256, 0, stream>>>(y, pM, pS, pL, tokLogp);
  cpo_finalize<<<1, 64, 0, stream>>>(y, tokLogp, out);
}
